// EpisodicMemory_19473381720682
// MI455X (gfx1250) — compile-verified, matrix operand order checked by reading, not run on hardware
//
#include <hip/hip_runtime.h>
#include <hip/hip_bf16.h>

typedef __attribute__((ext_vector_type(16))) _Float16 v16h;
typedef __attribute__((ext_vector_type(8)))  float    v8f;

#define B_   4
#define H_   16
#define S_   8192
#define HD_  64
#define D_   1024      // H_*HD_
#define E_   65536
#define KT_  10
#define L_   (KT_ + S_)
#define EPS_ 1e-8f

// ---------------------------------------------------------------------------
// 1) surprise[b,s] = || kf[b,s,:] - kf[b,s-1,:] ||   (kf = head-interleaved k)
__global__ void k_surprise(const float* __restrict__ k, float* __restrict__ sur) {
    int blk = blockIdx.x;
    int b = blk / S_, s = blk % S_;
    int t = threadIdx.x;
    __shared__ float red[256];
    float sum = 0.f;
    if (s > 0) {
        int d0 = t * 4;                 // 4 consecutive d, never crossing a head
        int h = d0 >> 6, hd = d0 & 63;
        size_t base = ((size_t)(b * H_ + h) * S_ + s) * HD_ + hd;
        float4 a = *(const float4*)(k + base);
        float4 p = *(const float4*)(k + base - HD_);   // s-1 is one HD_ stride back
        float dx = a.x - p.x, dy = a.y - p.y, dz = a.z - p.z, dw = a.w - p.w;
        sum = dx * dx + dy * dy + dz * dz + dw * dw;
    }
    red[t] = sum; __syncthreads();
    for (int o = 128; o > 0; o >>= 1) { if (t < o) red[t] += red[t + o]; __syncthreads(); }
    if (t == 0) sur[blk] = (s > 0) ? sqrtf(red[0]) : 0.f;
}

// ---------------------------------------------------------------------------
// 2) per-batch threshold = mean + 0.5 * std(ddof=1)
__global__ void k_stats(const float* __restrict__ sur, float* __restrict__ thr) {
    int b = blockIdx.x, t = threadIdx.x;
    __shared__ float rs[256], rq[256];
    float s1 = 0.f, s2 = 0.f;
    for (int i = t; i < S_; i += 256) { float x = sur[b * S_ + i]; s1 += x; s2 += x * x; }
    rs[t] = s1; rq[t] = s2; __syncthreads();
    for (int o = 128; o > 0; o >>= 1) {
        if (t < o) { rs[t] += rs[t + o]; rq[t] += rq[t + o]; }
        __syncthreads();
    }
    if (t == 0) {
        float mean = rs[0] / (float)S_;
        float var  = (rq[0] - rs[0] * rs[0] / (float)S_) / (float)(S_ - 1);
        thr[b] = mean + 0.5f * sqrtf(fmaxf(var, 0.f));
    }
}

// ---------------------------------------------------------------------------
// 3) boundary mask + exclusive scan -> seg_start[], nseg[]
__global__ void k_seg(const float* __restrict__ sur, const float* __restrict__ thr,
                      int* __restrict__ seg_start, int* __restrict__ nseg) {
    int b = blockIdx.x, t = threadIdx.x;   // 1024 threads, 8 elems each
    float T = thr[b];
    const float* sb = sur + b * S_;
    int bi[8], csum = 0;
    int s0 = t * 8;
    #pragma unroll
    for (int i = 0; i < 8; i++) {
        int s = s0 + i;
        bi[i] = ((sb[s] > T) || (s == S_ - 1)) ? 1 : 0;
        csum += bi[i];
    }
    __shared__ int sd[1024];
    sd[t] = csum; __syncthreads();
    for (int o = 1; o < 1024; o <<= 1) {
        int v = (t >= o) ? sd[t - o] : 0;
        __syncthreads();
        sd[t] += v;
        __syncthreads();
    }
    int excl = sd[t] - csum;               // boundaries strictly before s0
    int prev = 0;
    if (s0 > 0) prev = (sb[s0 - 1] > T) ? 1 : 0;   // s0-1 can never be S_-1
    int segid = excl;
    #pragma unroll
    for (int i = 0; i < 8; i++) {
        int s = s0 + i;
        int prev_bi = (i == 0) ? prev : bi[i - 1];
        if (s == 0 || prev_bi) seg_start[b * S_ + segid] = s;
        segid += bi[i];
    }
    if (t == 1023) nseg[b] = sd[1023];     // total boundaries == #segments
}

// ---------------------------------------------------------------------------
// 4) segment means -> new_events (B,S,D), event_lengths (B,S)
__global__ void k_segmean(const float* __restrict__ k, const int* __restrict__ seg_start,
                          const int* __restrict__ nseg,
                          float* __restrict__ new_events, float* __restrict__ ev_len) {
    int blk = blockIdx.x;
    int b = blk / S_, j = blk % S_;
    int t = threadIdx.x;
    int ns = nseg[b];
    float* outrow = new_events + ((size_t)b * S_ + j) * D_;
    if (j >= ns) {
        for (int d = t; d < D_; d += 256) outrow[d] = 0.f;
        if (t == 0) ev_len[b * S_ + j] = 0.f;
        return;
    }
    int start = seg_start[b * S_ + j];
    int end   = (j + 1 < ns) ? seg_start[b * S_ + j + 1] : S_;
    int cnt   = end - start;
    float inv = 1.f / (float)cnt;
    for (int d = t; d < D_; d += 256) {
        int h = d >> 6, hd = d & 63;
        const float* col = k + (size_t)(b * H_ + h) * S_ * HD_ + hd;
        float sum = 0.f;
        for (int s = start; s < end; s++) sum += col[(size_t)s * HD_];
        outrow[d] = sum * inv;
    }
    if (t == 0) ev_len[b * S_ + j] = (float)cnt;
}

// ---------------------------------------------------------------------------
// 5) query rows (padded to 16, f16) + 1/max(||query||, eps)
__global__ void k_query(const float* __restrict__ k, _Float16* __restrict__ qh,
                        float* __restrict__ inv_qn) {
    int m = blockIdx.x;                    // 0..15 (rows 4..15 zero)
    int t = threadIdx.x;
    __shared__ float red[256];
    if (m >= B_) {
        for (int d = t; d < D_; d += 256) qh[m * D_ + d] = (_Float16)0.f;
        return;
    }
    float s = 0.f;
    for (int d = t; d < D_; d += 256) {
        int h = d >> 6, hd = d & 63;
        float v = k[((size_t)(m * H_ + h) * S_ + (S_ - 1)) * HD_ + hd];
        qh[m * D_ + d] = (_Float16)v;
        s += v * v;
    }
    red[t] = s; __syncthreads();
    for (int o = 128; o > 0; o >>= 1) { if (t < o) red[t] += red[t + o]; __syncthreads(); }
    if (t == 0) inv_qn[m] = 1.f / fmaxf(sqrtf(red[0]), EPS_);
}

// ---------------------------------------------------------------------------
// 6) cosine sims via WMMA, with event row-norms fused into the same pass.
//    One wave computes a 16(batch-pad) x 16(events) tile; K=1024 in steps of 32
//    using v_wmma_f32_16x16x32_f16 (f32 accumulate). Each event row is read
//    exactly once (split across lane n / lane n^16); the f32 sum-of-squares is
//    accumulated alongside and combined with a ds_swizzle SWAPX16, so the
//    separate 256MB norm pass over `events` is eliminated.
__global__ void k_sims_wmma(const float* __restrict__ ev, const _Float16* __restrict__ qh,
                            const float* __restrict__ inv_qn,
                            float* __restrict__ sims) {
    int lane  = threadIdx.x;               // 32 lanes, one wave per block
    int ebase = blockIdx.x * 16;
    int nm = lane & 15;                    // == M for A-frag, == N for B-frag
    int hi = lane >> 4;
    const float2*   brow = (const float2*)(ev + (size_t)(ebase + nm) * D_);
    const _Float16* arow = qh + nm * D_;
    v8f c = {};
    float lsq = 0.f;                       // partial ||events row||^2 (this lane's K's)
    for (int k0 = 0; k0 < D_; k0 += 32) {
        v16h a, bm;
        #pragma unroll
        for (int v = 0; v < 8; v++) {
            // 16-bit A/B fragment layout: VGPR v holds K pair
            int K = ((v < 4) ? (v * 2) : (16 + (v - 4) * 2)) + hi * 8 + k0;
            a[2 * v]     = arow[K];
            a[2 * v + 1] = arow[K + 1];
            float2 w = brow[K >> 1];       // K is even -> aligned float2
            lsq += w.x * w.x + w.y * w.y;
            bm[2 * v]     = (_Float16)w.x;
            bm[2 * v + 1] = (_Float16)w.y;
        }
        c = __builtin_amdgcn_wmma_f32_16x16x32_f16(false, a, false, bm,
                                                   (short)0, c, false, false);
    }
    // combine lane n with lane n^16: ds_swizzle group-of-32 SWAPX16
    // offset = {xor=0x10, or=0x00, and=0x1f} -> 0x401F
    float osq = __int_as_float(
        __builtin_amdgcn_ds_swizzle(__float_as_int(lsq), 0x401F));
    float rowsq = lsq + osq;
    // D layout: lanes 0-15 hold N=lane, M = VGPR index r (0..7); rows 0..3 valid
    if (lane < 16) {
        int e = ebase + lane;
        float se = 1.f / fmaxf(sqrtf(rowsq), EPS_);
        #pragma unroll
        for (int r = 0; r < 4; r++)
            sims[(size_t)r * E_ + e] = c[r] * inv_qn[r] * se;
    }
}

// ---------------------------------------------------------------------------
// 7) top-10 per batch (iterative argmax; ties -> lowest index, matches top_k)
__global__ void k_topk(const float* __restrict__ sims, int* __restrict__ idx) {
    int b = blockIdx.x, t = threadIdx.x;
    __shared__ float bv[256];
    __shared__ int   bidx[256];
    __shared__ int   chosen[KT_];
    const float* sb = sims + (size_t)b * E_;
    for (int j = 0; j < KT_; j++) {
        float best = -1e30f; int besti = E_;
        for (int e = t; e < E_; e += 256) {
            bool skip = false;
            for (int q = 0; q < j; q++) if (chosen[q] == e) skip = true;
            if (skip) continue;
            float v = sb[e];
            if (v > best || (v == best && e < besti)) { best = v; besti = e; }
        }
        bv[t] = best; bidx[t] = besti; __syncthreads();
        for (int o = 128; o > 0; o >>= 1) {
            if (t < o) {
                if (bv[t + o] > bv[t] || (bv[t + o] == bv[t] && bidx[t + o] < bidx[t])) {
                    bv[t] = bv[t + o]; bidx[t] = bidx[t + o];
                }
            }
            __syncthreads();
        }
        if (t == 0) { chosen[j] = bidx[0]; idx[b * KT_ + j] = bidx[0]; }
        __syncthreads();
    }
}

// ---------------------------------------------------------------------------
// 8) aug_k / aug_v assembly (float4 vectorized single pass)
__global__ void k_aug(const float* __restrict__ kk, const float* __restrict__ vv,
                      const float* __restrict__ ev, const int* __restrict__ idx,
                      float* __restrict__ aug_k, float* __restrict__ aug_v) {
    size_t i = (size_t)blockIdx.x * blockDim.x + threadIdx.x;   // float4 index
    const size_t total = (size_t)B_ * H_ * L_ * HD_ / 4;
    if (i >= total) return;
    int    hd4 = (int)(i & 15);
    size_t r   = i >> 4;
    int    p   = (int)(r % L_);
    size_t r2  = r / L_;
    int    h   = (int)(r2 & 15);
    int    b   = (int)(r2 >> 4);
    float4 kq, vq;
    if (p < KT_) {
        const float4* src = (const float4*)(ev + (size_t)idx[b * KT_ + p] * D_ + h * HD_);
        kq = src[hd4]; vq = kq;
    } else {
        size_t base = ((size_t)(b * H_ + h) * S_ + (size_t)(p - KT_)) * HD_;
        kq = ((const float4*)(kk + base))[hd4];
        vq = ((const float4*)(vv + base))[hd4];
    }
    ((float4*)aug_k)[i] = kq;
    ((float4*)aug_v)[i] = vq;
}

// ---------------------------------------------------------------------------
// 9) aug_mask + aug_positions
__global__ void k_maskpos(const float* __restrict__ am,
                          float* __restrict__ aug_mask, float* __restrict__ aug_pos) {
    int i = blockIdx.x * blockDim.x + threadIdx.x;
    if (i >= B_ * L_) return;
    int b = i / L_, p = i % L_;
    aug_mask[i] = (p < KT_) ? 1.f : am[b * S_ + p - KT_];
    aug_pos[i]  = (p < KT_) ? (float)(S_ - 1 - KT_ + p) : (float)(p - KT_);
}

// ---------------------------------------------------------------------------
extern "C" void kernel_launch(void* const* d_in, const int* in_sizes, int n_in,
                              void* d_out, int out_size, void* d_ws, size_t ws_size,
                              hipStream_t stream) {
    const float* inp = (const float*)d_in[0];
    const float* q   = (const float*)d_in[1];
    const float* k   = (const float*)d_in[2];
    const float* v   = (const float*)d_in[3];
    const float* am  = (const float*)d_in[4];
    const float* ev  = (const float*)d_in[5];
    float* out = (float*)d_out;

    // output offsets (floats), reference return order
    const size_t sz_big = (size_t)B_ * S_ * D_;           // 33,554,432
    const size_t sz_aug = (size_t)B_ * H_ * L_ * HD_;     // 33,595,392
    const size_t o_inputs = 0;
    const size_t o_q      = o_inputs + sz_big;
    const size_t o_ak     = o_q + sz_big;
    const size_t o_av     = o_ak + sz_aug;
    const size_t o_mask   = o_av + sz_aug;
    const size_t o_pos    = o_mask + (size_t)B_ * L_;
    const size_t o_ne     = o_pos + (size_t)B_ * L_;
    const size_t o_el     = o_ne + sz_big;

    // workspace carve-up (~1.4 MB)
    char* w = (char*)d_ws;
    float* sur       = (float*)w;  w += sizeof(float) * B_ * S_;
    float* thr       = (float*)w;  w += sizeof(float) * 16;
    int*   seg_start = (int*)w;    w += sizeof(int) * B_ * S_;
    int*   nseg      = (int*)w;    w += sizeof(int) * 16;
    float* inv_qn    = (float*)w;  w += sizeof(float) * 16;
    _Float16* qh     = (_Float16*)w; w += sizeof(_Float16) * 16 * D_;
    float* sims      = (float*)w;  w += sizeof(float) * (size_t)B_ * E_;
    int*   idx       = (int*)w;    w += sizeof(int) * B_ * KT_;

    // passthroughs
    hipMemcpyAsync(out + o_inputs, inp, sz_big * sizeof(float),
                   hipMemcpyDeviceToDevice, stream);
    hipMemcpyAsync(out + o_q, q, sz_big * sizeof(float),
                   hipMemcpyDeviceToDevice, stream);

    k_surprise<<<B_ * S_, 256, 0, stream>>>(k, sur);
    k_stats  <<<B_, 256, 0, stream>>>(sur, thr);
    k_seg    <<<B_, 1024, 0, stream>>>(sur, thr, seg_start, nseg);
    k_segmean<<<B_ * S_, 256, 0, stream>>>(k, seg_start, nseg, out + o_ne, out + o_el);
    k_query  <<<16, 256, 0, stream>>>(k, qh, inv_qn);
    k_sims_wmma<<<E_ / 16, 32, 0, stream>>>(ev, qh, inv_qn, sims);
    k_topk   <<<B_, 256, 0, stream>>>(sims, idx);
    {
        const size_t total4 = (size_t)B_ * H_ * L_ * HD_ / 4;
        k_aug<<<(unsigned)((total4 + 255) / 256), 256, 0, stream>>>(
            k, v, ev, idx, out + o_ak, out + o_av);
    }
    k_maskpos<<<(B_ * L_ + 255) / 256, 256, 0, stream>>>(am, out + o_mask, out + o_pos);
}
